// NBRCLayer_88587995447899
// MI455X (gfx1250) — matrix operand order repaired and checked by reading, not compiled, not run on hardware
//
#include <hip/hip_runtime.h>

// ---------------------------------------------------------------------------
// nBRC layer on gfx1250 (MI455X): bf16 WMMA GEMMs with LDS-staged B operands
// + fused recurrence.
// ---------------------------------------------------------------------------

typedef __attribute__((ext_vector_type(16))) __bf16 v16bf;
typedef __attribute__((ext_vector_type(8)))  __bf16 v8bf;
typedef __attribute__((ext_vector_type(4)))  __bf16 v4bf;
typedef __attribute__((ext_vector_type(8)))  float  v8f;

namespace {
constexpr int kT = 512;   // time steps
constexpr int kB = 256;   // batch
constexpr int kI = 512;   // input dim
constexpr int kH = 1024;  // hidden dim
}

// ---- WMMA fragment helpers (ISA 7.12.2 bf16 layouts, wave32) --------------
// A 16x32 (MxK): lane L<16 -> row L, K = k0+0..7 (v0-3) and k0+16..23 (v4-7)
//                lane L>=16 -> row L-16, K = k0+8..15 and k0+24..31
__device__ __forceinline__ v16bf load_a_frag(const __bf16* __restrict__ tile,
                                             int ldk, int lane) {
  const int r    = lane & 15;
  const int dsel = (lane >> 4) << 3;  // 0 or 8
  const __bf16* p = tile + (size_t)r * ldk + dsel;
  v8bf lo = *reinterpret_cast<const v8bf*>(p);       // K = d+0..7
  v8bf hi = *reinterpret_cast<const v8bf*>(p + 16);  // K = d+16..23
  return __builtin_shufflevector(lo, hi, 0, 1, 2, 3, 4, 5, 6, 7,
                                 8, 9, 10, 11, 12, 13, 14, 15);
}

// B 32x16 (KxN), sourced from Bt[N, K] (row n = output column, K-contiguous):
// lane L<16 -> col L, K = k0+0..7,16..23 ; lane L>=16 -> col L-16, K = k0+8..15,24..31
__device__ __forceinline__ v16bf load_b_frag(const __bf16* __restrict__ tile,
                                             int ldk, int lane) {
  const int n  = lane & 15;
  const int ks = (lane >> 4) << 3;  // 0 or 8
  const __bf16* p = tile + (size_t)n * ldk + ks;
  v8bf lo = *reinterpret_cast<const v8bf*>(p);
  v8bf hi = *reinterpret_cast<const v8bf*>(p + 16);
  return __builtin_shufflevector(lo, hi, 0, 1, 2, 3, 4, 5, 6, 7,
                                 8, 9, 10, 11, 12, 13, 14, 15);
}

__device__ __forceinline__ v8f wmma_bf16(v16bf a, v16bf b, v8f c) {
  return __builtin_amdgcn_wmma_f32_16x16x32_bf16(
      /*neg_a=*/false, a, /*neg_b=*/false, b,
      /*c_mod=*/(short)0, c, /*reuse_a=*/false, /*reuse_b=*/false);
}

__device__ __forceinline__ float sigmoidf_dev(float x) {
  return 1.0f / (1.0f + __expf(-x));
}

// ---- f32 -> bf16 conversion (4-wide) --------------------------------------
__global__ __launch_bounds__(256) void cvt_f32_bf16(
    const float* __restrict__ src, __bf16* __restrict__ dst, long n4) {
  long i = (long)blockIdx.x * blockDim.x + threadIdx.x;
  if (i >= n4) return;
  const float4 v = reinterpret_cast<const float4*>(src)[i];
  v4bf o;
  o[0] = (__bf16)v.x; o[1] = (__bf16)v.y;
  o[2] = (__bf16)v.z; o[3] = (__bf16)v.w;
  reinterpret_cast<v4bf*>(dst)[i] = o;
}

// ---- Phase 1: xc/xa/xh = x @ U{c,a,h}^T (+bias) ---------------------------
// Block = 8 waves covering 128 rows x 64-col strip; U tiles (all three
// matrices) staged through LDS in K-chunks of 64 and shared by the 8 waves.
__global__ __launch_bounds__(256) void proj_kernel(
    const __bf16* __restrict__ xb,                       // [M=T*B, I]
    const __bf16* __restrict__ Ucb,
    const __bf16* __restrict__ Uab,
    const __bf16* __restrict__ Uhb,                      // [H, I]
    const float* __restrict__ b_c, const float* __restrict__ b_a,
    float* __restrict__ xc, float* __restrict__ xa,
    float* __restrict__ xh)                              // [M, H] each
{
  constexpr int kLdB = 64 + 8;                       // bank-padded row stride
  __shared__ __bf16 Bs[3][64][kLdB];                 // 3 x 64 x 64 K-chunk

  const int tid  = threadIdx.x;
  const int lane = tid & 31;
  const int w    = tid >> 5;                         // wave in block (0..7)
  const int ns   = blockIdx.x & 15;                  // N strip (kH/64)
  const int mg   = blockIdx.x >> 4;                  // 128-row group
  const int m0   = mg * 128 + w * 16;
  const int n0   = ns * 64;

  const __bf16* Us[3] = {Ucb, Uab, Uhb};

  v8f accC[4] = {}, accA[4] = {}, accH[4] = {};
  const __bf16* aBase = xb + (size_t)m0 * kI;

  for (int k0 = 0; k0 < kI; k0 += 64) {
    __syncthreads();
    // Stage 3 x 64rows x 64K bf16 tiles: 512 16B-chunks per matrix.
#pragma unroll
    for (int m = 0; m < 3; ++m) {
      const __bf16* src = Us[m];
#pragma unroll
      for (int i = 0; i < 2; ++i) {
        const int c   = tid + i * 256;               // 0..511
        const int row = c >> 3;                      // 0..63
        const int off = (c & 7) * 8;                 // 0..56 (16B chunks)
        const __bf16* gp = src + (size_t)(n0 + row) * kI + k0 + off;
        *reinterpret_cast<v8bf*>(&Bs[m][row][off]) =
            *reinterpret_cast<const v8bf*>(gp);
        if (k0 + 64 < kI) __builtin_prefetch(gp + 64, 0, 1);
      }
    }
    __syncthreads();

#pragma unroll
    for (int kk = 0; kk < 64; kk += 32) {
      const v16bf af = load_a_frag(aBase + k0 + kk, kI, lane);
#pragma unroll
      for (int j = 0; j < 4; ++j) {
        accC[j] = wmma_bf16(af, load_b_frag(&Bs[0][j * 16][kk], kLdB, lane),
                            accC[j]);
        accA[j] = wmma_bf16(af, load_b_frag(&Bs[1][j * 16][kk], kLdB, lane),
                            accA[j]);
        accH[j] = wmma_bf16(af, load_b_frag(&Bs[2][j * 16][kk], kLdB, lane),
                            accH[j]);
      }
    }
  }

  // C/D layout: VGPR r, lane l -> M = r + 8*(l>>4), N = l&15
  const int nl = lane & 15;
  const int mo = (lane >> 4) << 3;
#pragma unroll
  for (int j = 0; j < 4; ++j) {
    const int n = n0 + j * 16 + nl;
    const float bc = b_c[n];
    const float ba = b_a[n];
#pragma unroll
    for (int r = 0; r < 8; ++r) {
      const size_t idx = (size_t)(m0 + mo + r) * kH + n;
      xc[idx] = accC[j][r] + bc;
      xa[idx] = accA[j][r] + ba;
      xh[idx] = accH[j][r];
    }
  }
}

// ---- Phase 2: one recurrence step -----------------------------------------
// c = sigmoid(xc_t + h W_c^T); a = 1 + tanh(xa_t + h W_a^T)
// h' = c*h + (1-c)*tanh(xh_t + a*h)   (xh_t lives in out_t, updated in place)
// Block = 8 waves covering 128 rows x 32-col strip; W_c/W_a tiles staged
// through LDS in K-chunks of 128 and shared by the 8 waves. 64 blocks/step.
__global__ __launch_bounds__(256) void step_kernel(
    const __bf16* __restrict__ hpb,                      // h_prev bf16 [B,H]
    const float* __restrict__ hpf,                       // h_prev f32  [B,H]
    const __bf16* __restrict__ Wcb,
    const __bf16* __restrict__ Wab,                      // [H, H]
    const float* __restrict__ xc_t,
    const float* __restrict__ xa_t,                      // [B, H]
    float* __restrict__ out_t,                           // xh in, h_new out
    __bf16* __restrict__ hnb)                            // h_new bf16 [B,H]
{
  constexpr int kLdB = 128 + 8;                      // bank-padded row stride
  __shared__ __bf16 Bs[2][32][kLdB];                 // {W_c,W_a} 32 x 128 K

  const int tid  = threadIdx.x;
  const int lane = tid & 31;
  const int w    = tid >> 5;                         // wave in block (0..7)
  const int cs   = blockIdx.x & 31;                  // 32-col strip (kH/32)
  const int mg   = blockIdx.x >> 5;                  // 128-row group (0..1)
  const int m0   = mg * 128 + w * 16;
  const int n0   = cs * 32;

  v8f accC[2] = {}, accA[2] = {};
  const __bf16* aBase = hpb + (size_t)m0 * kH;

  for (int k0 = 0; k0 < kH; k0 += 128) {
    __syncthreads();
    // Stage 2 x 32rows x 128K bf16 tiles: 1024 16B-chunks total.
#pragma unroll
    for (int i = 0; i < 4; ++i) {
      const int c   = tid + i * 256;                 // 0..1023
      const int m   = c >> 9;                        // 0..1
      const int cc  = c & 511;
      const int row = cc >> 4;                       // 0..31
      const int off = (cc & 15) * 8;                 // 0..120
      const __bf16* src = (m == 0) ? Wcb : Wab;
      const __bf16* gp  = src + (size_t)(n0 + row) * kH + k0 + off;
      *reinterpret_cast<v8bf*>(&Bs[m][row][off]) =
          *reinterpret_cast<const v8bf*>(gp);
      if (k0 + 128 < kH) __builtin_prefetch(gp + 128, 0, 1);
    }
    __syncthreads();

#pragma unroll
    for (int kk = 0; kk < 128; kk += 32) {
      const v16bf af = load_a_frag(aBase + k0 + kk, kH, lane);
#pragma unroll
      for (int j = 0; j < 2; ++j) {
        accC[j] = wmma_bf16(af, load_b_frag(&Bs[0][j * 16][kk], kLdB, lane),
                            accC[j]);
        accA[j] = wmma_bf16(af, load_b_frag(&Bs[1][j * 16][kk], kLdB, lane),
                            accA[j]);
      }
    }
  }

  const int nl = lane & 15;
  const int mo = (lane >> 4) << 3;
#pragma unroll
  for (int j = 0; j < 2; ++j) {
#pragma unroll
    for (int r = 0; r < 8; ++r) {
      const size_t idx = (size_t)(m0 + mo + r) * kH + (n0 + j * 16 + nl);
      const float c  = sigmoidf_dev(accC[j][r] + xc_t[idx]);
      const float a  = 1.0f + tanhf(accA[j][r] + xa_t[idx]);
      const float hp = hpf[idx];
      const float hn = c * hp + (1.0f - c) * tanhf(out_t[idx] + a * hp);
      out_t[idx] = hn;
      hnb[idx]   = (__bf16)hn;
    }
  }
}

// ---------------------------------------------------------------------------
extern "C" void kernel_launch(void* const* d_in, const int* in_sizes, int n_in,
                              void* d_out, int out_size, void* d_ws,
                              size_t ws_size, hipStream_t stream) {
  (void)in_sizes; (void)n_in; (void)out_size; (void)ws_size;

  const float* x_seq = (const float*)d_in[0];   // [T,B,I]
  const float* h0    = (const float*)d_in[1];   // [B,H]
  const float* U_c   = (const float*)d_in[2];   // [H,I]
  const float* W_c   = (const float*)d_in[3];   // [H,H]
  const float* b_c   = (const float*)d_in[4];   // [H]
  const float* U_a   = (const float*)d_in[5];   // [H,I]
  const float* W_a   = (const float*)d_in[6];   // [H,H]
  const float* b_a   = (const float*)d_in[7];   // [H]
  const float* U_h   = (const float*)d_in[8];   // [H,I]
  float* out = (float*)d_out;                   // [T,B,H] ++ [B,H]

  // Workspace carve-up.
  char* ws = (char*)d_ws;
  size_t off = 0;
  auto carve = [&](size_t bytes) -> char* {
    char* p = ws + off;
    off += (bytes + 255) & ~(size_t)255;
    return p;
  };
  __bf16* xb  = (__bf16*)carve((size_t)kT * kB * kI * 2);
  __bf16* Ucb = (__bf16*)carve((size_t)kH * kI * 2);
  __bf16* Uab = (__bf16*)carve((size_t)kH * kI * 2);
  __bf16* Uhb = (__bf16*)carve((size_t)kH * kI * 2);
  __bf16* Wcb = (__bf16*)carve((size_t)kH * kH * 2);
  __bf16* Wab = (__bf16*)carve((size_t)kH * kH * 2);
  __bf16* hb0 = (__bf16*)carve((size_t)kB * kH * 2);
  __bf16* hb1 = (__bf16*)carve((size_t)kB * kH * 2);
  float*  xc  = (float*)carve((size_t)kT * kB * kH * 4);
  float*  xa  = (float*)carve((size_t)kT * kB * kH * 4);

  auto cvt = [&](const float* s, __bf16* d, size_t n) {
    const long n4 = (long)(n / 4);
    const long nb = (n4 + 255) / 256;
    cvt_f32_bf16<<<dim3((unsigned)nb), dim3(256), 0, stream>>>(s, d, n4);
  };

  // Phase 0: bf16 staging of operands.
  cvt(x_seq, xb,  (size_t)kT * kB * kI);
  cvt(U_c,   Ucb, (size_t)kH * kI);
  cvt(U_a,   Uab, (size_t)kH * kI);
  cvt(U_h,   Uhb, (size_t)kH * kI);
  cvt(W_c,   Wcb, (size_t)kH * kH);
  cvt(W_a,   Wab, (size_t)kH * kH);
  cvt(h0,    hb0, (size_t)kB * kH);

  // Phase 1: input projections. Block = 128 rows x 64 cols.
  {
    const unsigned blocks = (unsigned)((kT * kB / 128) * (kH / 64));  // 16384
    proj_kernel<<<dim3(blocks), dim3(256), 0, stream>>>(
        xb, Ucb, Uab, Uhb, b_c, b_a, xc, xa, out);
  }

  // Phase 2: sequential recurrence, one kernel per step.
  {
    const unsigned blocks = (unsigned)((kB / 128) * (kH / 32));       // 64
    for (int t = 0; t < kT; ++t) {
      const __bf16* hpb = (t & 1) ? hb1 : hb0;
      __bf16*       hnb = (t & 1) ? hb0 : hb1;
      const float*  hpf = (t == 0) ? h0 : out + (size_t)(t - 1) * kB * kH;
      step_kernel<<<dim3(blocks), dim3(256), 0, stream>>>(
          hpb, hpf, Wcb, Wab,
          xc + (size_t)t * kB * kH, xa + (size_t)t * kB * kH,
          out + (size_t)t * kB * kH, hnb);
    }
  }

  // hn = output[T-1]
  hipMemcpyAsync(out + (size_t)kT * kB * kH, out + (size_t)(kT - 1) * kB * kH,
                 (size_t)kB * kH * sizeof(float), hipMemcpyDeviceToDevice,
                 stream);
}
